// RWKV7_TimeMix_63539746177181
// MI455X (gfx1250) — matrix operand order repaired and checked by reading, not compiled, not run on hardware
//
#include <hip/hip_runtime.h>
#include <hip/hip_bf16.h>
#include <math.h>
#include <stdint.h>

#define Bdim 2
#define Tdim 2048
#define Cdim 1024
#define Hdim 16
#define Ndim 64
#define Mrows (Bdim * Tdim)

typedef unsigned short bf16_t;
typedef __bf16 bf16x16 __attribute__((ext_vector_type(16)));
typedef float f32x8 __attribute__((ext_vector_type(8)));

// CDNA5 async global->LDS copy (ASYNCcnt-tracked), with safe fallback.
// Builtin signature (from hipcc diagnostics): (int4 AS1*, int4 AS3*, Ii, Ii).
#if __has_builtin(__builtin_amdgcn_global_load_async_to_lds_b128) && \
    __has_builtin(__builtin_amdgcn_s_wait_asynccnt)
#define HAVE_ASYNC_LDS 1
typedef int v4i_t __attribute__((vector_size(4 * sizeof(int))));
typedef __attribute__((address_space(1))) v4i_t as1_v4i;
typedef __attribute__((address_space(3))) v4i_t as3_v4i;
#define ASYNC_CP_B128(g, l)                                                   \
  __builtin_amdgcn_global_load_async_to_lds_b128(                             \
      (as1_v4i*)(uintptr_t)(g),                                               \
      (as3_v4i*)(uint32_t)(uintptr_t)(l), 0, 0)
#else
#define HAVE_ASYNC_LDS 0
#endif

__device__ __forceinline__ bf16_t f2bf(float f) {
  unsigned int u = __float_as_uint(f);
  u += 0x7FFFu + ((u >> 16) & 1u);          // round-to-nearest-even
  return (bf16_t)(u >> 16);
}
__device__ __forceinline__ float sigm(float z) { return 1.0f / (1.0f + expf(-z)); }

// ---------------- elementwise kernels ----------------

// Pack weight matrix B[K,N] (f32, row-major) into bf16 WMMA-fragment order:
// dst[((ct*KT + kt)*32 + lane)*16 + q]  where
//   ct = column tile (16 cols), kt = K tile (32 rows), KT = K/32,
//   lane: col = ct*16 + (lane&15), k = (lane>>4)*8 + (q&7) + (q>>3)*16 (A-frag order).
// In the GEMM each lane then reads its 16-value B fragment as two b128 loads.
__global__ void pack_b_kernel(const float* __restrict__ src, bf16_t* __restrict__ dst,
                              int N, int K) {
  int e = blockIdx.x * blockDim.x + threadIdx.x;
  if (e >= N * K) return;
  int q = e & 15;
  int lane = (e >> 4) & 31;
  int KT = K >> 5;
  int tile = e >> 9;
  int kt = tile % KT;
  int ct = tile / KT;
  int col = ct * 16 + (lane & 15);
  int krow = kt * 32 + (lane >> 4) * 8 + (q & 7) + (q >> 3) * 16;
  dst[e] = f2bf(src[(size_t)krow * N + col]);
}

// time_shift mixing: out_i = x + (shift(x) - x) * mix_i, 6 outputs, bf16
__global__ void mix6_kernel(const float* __restrict__ x,
                            const float* __restrict__ mr, const float* __restrict__ mw,
                            const float* __restrict__ mk, const float* __restrict__ mv,
                            const float* __restrict__ ma, const float* __restrict__ mg,
                            bf16_t* __restrict__ xr, bf16_t* __restrict__ xw,
                            bf16_t* __restrict__ xk, bf16_t* __restrict__ xv,
                            bf16_t* __restrict__ xa, bf16_t* __restrict__ xg) {
  int i = blockIdx.x * blockDim.x + threadIdx.x;   // < M*C
  int c = i & (Cdim - 1);
  int row = i >> 10;
  int t = row & (Tdim - 1);
  float xc = x[i];
  float xp = (t == 0) ? 0.0f : x[i - Cdim];
  float d = xp - xc;
  xr[i] = f2bf(xc + d * mr[c]);
  xw[i] = f2bf(xc + d * mw[c]);
  xk[i] = f2bf(xc + d * mk[c]);
  xv[i] = f2bf(xc + d * mv[c]);
  xa[i] = f2bf(xc + d * ma[c]);
  xg[i] = f2bf(xc + d * mg[c]);
}

// mode 0: tanh, 1: identity, 2: sigmoid  (f32 -> bf16)
__global__ void act_kernel(const float* __restrict__ s, bf16_t* __restrict__ d, int n, int mode) {
  int i = blockIdx.x * blockDim.x + threadIdx.x;
  if (i >= n) return;
  float v = s[i];
  if (mode == 0) v = tanhf(v);
  else if (mode == 2) v = sigm(v);
  d[i] = f2bf(v);
}

// per-element post processing after stage-2 GEMMs
__global__ void post_kernel(const float* __restrict__ w2o, const float* __restrict__ a2o,
                            const float* __restrict__ v2o, const float* __restrict__ vbuf,
                            const float* __restrict__ vfirst, const float* __restrict__ kbuf,
                            const float* __restrict__ w0, const float* __restrict__ a0,
                            const float* __restrict__ v0, const float* __restrict__ k_a,
                            float* __restrict__ decay, float* __restrict__ asig,
                            float* __restrict__ vnew, float* __restrict__ knew) {
  int i = blockIdx.x * blockDim.x + threadIdx.x;  // < M*C
  int c = i & (Cdim - 1);
  float u = w0[c] + w2o[i];
  float nu = -u;
  float sp = (nu > 20.0f) ? nu : log1pf(expf(nu));   // softplus(-u)
  float w = -sp - 0.5f;
  decay[i] = expf(-expf(w));                          // per-channel decay for the scan
  float a = sigm(a0[c] + a2o[i]);
  asig[i] = a;
  float sv = sigm(v0[c] + v2o[i]);
  float v = vbuf[i];
  vnew[i] = v + (vfirst[i] - v) * sv;
  knew[i] = kbuf[i] * (1.0f + (a - 1.0f) * k_a[c]);
}

// per-head l2norm of k*k_k -> kk
__global__ __launch_bounds__(64) void kk_kernel(const float* __restrict__ kbuf,
                                                const float* __restrict__ k_k,
                                                float* __restrict__ kkout) {
  __shared__ float red[64];
  int g = blockIdx.x;                 // row*H + h
  int row = g >> 4;
  int h = g & (Hdim - 1);
  int n = threadIdx.x;
  int c = h * Ndim + n;
  size_t idx = (size_t)row * Cdim + c;
  float e = kbuf[idx] * k_k[c];
  red[n] = e * e;
  __syncthreads();
  for (int s = 32; s > 0; s >>= 1) { if (n < s) red[n] += red[n + s]; __syncthreads(); }
  float inv = rsqrtf(red[0] + 1e-12f);
  inv = fmaxf(inv, 1e-12f);
  kkout[idx] = e * inv;
}

// ---------------- WMMA GEMM ----------------
// C[M,N] f32 = A[M,K] bf16 (row-major) @ Bp (fragment-packed bf16).
// Block = 256 threads = 8 waves; each wave computes a 16 x (16*NT) strip:
// one A fragment feeds NT back-to-back WMMAs (A reuse), all operands loaded
// as b128. A-frag layout per ISA 7.12.2 (16-bit A 16x32): lane lp=lane&15 is
// row, hl=lane>>4 selects K chunks [hl*8, hl*8+8) and [16+hl*8, ...).
template <int NT>
__global__ __launch_bounds__(256) void wmma_gemm_bf16(const bf16_t* __restrict__ A,
                                                      const bf16_t* __restrict__ Bp,
                                                      float* __restrict__ Cm,
                                                      int M, int N, int K) {
  int lane = threadIdx.x & 31;
  int wave = threadIdx.x >> 5;
  int tileRow = blockIdx.x * 128 + wave * 16;
  int ct0 = blockIdx.y * NT;
  int lp = lane & 15;
  int hl = lane >> 4;
  int KT = K >> 5;

  const bf16_t* arow = A + (size_t)(tileRow + lp) * K + hl * 8;

  f32x8 zero = {0.f, 0.f, 0.f, 0.f, 0.f, 0.f, 0.f, 0.f};
  f32x8 acc[NT];
#pragma unroll
  for (int n = 0; n < NT; ++n) acc[n] = zero;

  union Frag { uint4 u[2]; bf16x16 v; };

  for (int kt = 0; kt < KT; ++kt) {
    Frag af;
    af.u[0] = *(const uint4*)(arow + kt * 32);
    af.u[1] = *(const uint4*)(arow + kt * 32 + 16);
    __builtin_prefetch(arow + kt * 32 + 32, 0, 1);
#pragma unroll
    for (int n = 0; n < NT; ++n) {
      const uint4* bp = (const uint4*)(Bp + ((((size_t)(ct0 + n) * KT + kt) * 32 + lane) * 16));
      Frag bfr;
      bfr.u[0] = bp[0];
      bfr.u[1] = bp[1];
      acc[n] = __builtin_amdgcn_wmma_f32_16x16x32_bf16(false, af.v, false, bfr.v,
                                                       (short)0, acc[n], false, false);
    }
  }

  // C/D layout: VGPR v -> row (v + 8*hl), col = lane&15
#pragma unroll
  for (int n = 0; n < NT; ++n) {
    float* cp = Cm + (size_t)(tileRow + hl * 8) * N + (ct0 + n) * 16 + lp;
#pragma unroll
    for (int v = 0; v < 8; ++v) cp[(size_t)v * N] = acc[n][v];
  }
}

// ---------------- RWKV7 sequential scan ----------------
// one block per (b,h); 256 threads; thread (i=tid>>2, seg=tid&3) owns
// state[i][seg*16 .. seg*16+15]. 4-lane quad reductions via shfl_xor (wave32).
// Double-buffered: step t+1's six 64-float vectors stream into LDS (async
// global->LDS DMA on CDNA5, register-staged fallback otherwise) while step t
// computes, hiding the L2 load latency that would otherwise serialize into
// the 2048-step recurrence.
__global__ __launch_bounds__(256) void scan_kernel(const float* __restrict__ r,
                                                   const float* __restrict__ decay,
                                                   const float* __restrict__ knew,
                                                   const float* __restrict__ vnew,
                                                   const float* __restrict__ kk,
                                                   const float* __restrict__ asig,
                                                   float* __restrict__ y) {
  __shared__ __align__(16) float buf[2][6][64];   // [buffer][array][n]
  int bh = blockIdx.x;               // [0, B*H)
  int b = bh >> 4;
  int h = bh & (Hdim - 1);
  int tid = threadIdx.x;
  int i = tid >> 2;
  int seg = tid & 3;
  int j0 = seg * 16;
  size_t base = ((size_t)b * Tdim) * Cdim + (size_t)h * Ndim;

  // loader mapping: 96 lanes x 16B; which array, float4 slot
  int which = tid >> 4;              // 0..5 for tid < 96
  int n4 = (tid & 15) * 4;
  const bool ld = tid < 96;
  const float* sp = (which == 0) ? r
                  : (which == 1) ? decay
                  : (which == 2) ? knew
                  : (which == 3) ? vnew
                  : (which == 4) ? kk
                                 : asig;

  // prologue: fetch t = 0 into buf[0]
#if HAVE_ASYNC_LDS
  if (ld) ASYNC_CP_B128(sp + base + n4, &buf[0][which][n4]);
  __builtin_amdgcn_s_wait_asynccnt(0);
  __syncthreads();
#else
  float4 pf;
  if (ld) {
    pf = *(const float4*)(sp + base + n4);
    *(float4*)&buf[0][which][n4] = pf;
  }
  __syncthreads();
#endif

  float st[16];
#pragma unroll
  for (int q = 0; q < 16; ++q) st[q] = 0.0f;

  for (int t = 0; t < Tdim; ++t) {
    int cur = t & 1;
    int nxt = cur ^ 1;

    // kick off step t+1 fetch (overlaps with compute below)
    if (t + 1 < Tdim && ld) {
      const float* gp = sp + base + (size_t)(t + 1) * Cdim + n4;
#if HAVE_ASYNC_LDS
      ASYNC_CP_B128(gp, &buf[nxt][which][n4]);
#else
      pf = *(const float4*)gp;
#endif
    }

    const float* r_s  = buf[cur][0];
    const float* d_s  = buf[cur][1];
    const float* k_s  = buf[cur][2];
    const float* v_s  = buf[cur][3];
    const float* kk_s = buf[cur][4];
    const float* a_s  = buf[cur][5];

    // sa_i = sum_j state_ij * (-kk_j)
    float sa = 0.0f;
#pragma unroll
    for (int q = 0; q < 16; ++q) sa -= st[q] * kk_s[j0 + q];
    sa += __shfl_xor(sa, 1, 32);
    sa += __shfl_xor(sa, 2, 32);

    float vi = v_s[i];
    float out = 0.0f;
#pragma unroll
    for (int q = 0; q < 16; ++q) {
      int j = j0 + q;
      float bj = kk_s[j] * a_s[j];
      float s2 = fmaf(st[q], d_s[j], fmaf(sa, bj, vi * k_s[j]));
      st[q] = s2;
      out = fmaf(s2, r_s[j], out);
    }
    out += __shfl_xor(out, 1, 32);
    out += __shfl_xor(out, 2, 32);
    if (seg == 0) y[base + (size_t)t * Cdim + i] = out;

#if HAVE_ASYNC_LDS
    __builtin_amdgcn_s_wait_asynccnt(0);
    __syncthreads();
#else
    if (t + 1 < Tdim && ld) *(float4*)&buf[nxt][which][n4] = pf;
    __syncthreads();
#endif
  }
}

// GroupNorm (per head) + rwkv residual + gate -> bf16 activation for final GEMM
__global__ __launch_bounds__(64) void gn_res_kernel(const float* __restrict__ y,
                                                    const float* __restrict__ r,
                                                    const float* __restrict__ knew,
                                                    const float* __restrict__ vnew,
                                                    const float* __restrict__ gbuf,
                                                    const float* __restrict__ r_k,
                                                    const float* __restrict__ gamma,
                                                    const float* __restrict__ beta,
                                                    bf16_t* __restrict__ act) {
  __shared__ float red[64];
  int g = blockIdx.x;                 // row*H + h
  int row = g >> 4;
  int h = g & (Hdim - 1);
  int n = threadIdx.x;
  int c = h * Ndim + n;
  size_t idx = (size_t)row * Cdim + c;

  float yv = y[idx];
  red[n] = yv;
  __syncthreads();
  for (int s = 32; s > 0; s >>= 1) { if (n < s) red[n] += red[n + s]; __syncthreads(); }
  float mu = red[0] * (1.0f / 64.0f);
  __syncthreads();

  float dv = yv - mu;
  red[n] = dv * dv;
  __syncthreads();
  for (int s = 32; s > 0; s >>= 1) { if (n < s) red[n] += red[n + s]; __syncthreads(); }
  float var = red[0] * (1.0f / 64.0f);
  __syncthreads();

  float gn = dv * rsqrtf(var + 0.00064f) * gamma[c] + beta[c];

  red[n] = r[idx] * knew[idx] * r_k[c];   // r_k flat index == channel index
  __syncthreads();
  for (int s = 32; s > 0; s >>= 1) { if (n < s) red[n] += red[n + s]; __syncthreads(); }
  float srk = red[0];

  float yfin = gn + srk * vnew[idx];
  act[idx] = f2bf(yfin * gbuf[idx]);
}

// ---------------- host launch ----------------

extern "C" void kernel_launch(void* const* d_in, const int* in_sizes, int n_in,
                              void* d_out, int out_size, void* d_ws, size_t ws_size,
                              hipStream_t stream) {
  const float* x       = (const float*)d_in[0];
  const float* v_first = (const float*)d_in[1];
  const float* x_r = (const float*)d_in[2];
  const float* x_w = (const float*)d_in[3];
  const float* x_k = (const float*)d_in[4];
  const float* x_v = (const float*)d_in[5];
  const float* x_a = (const float*)d_in[6];
  const float* x_g = (const float*)d_in[7];
  const float* w0 = (const float*)d_in[8];
  const float* w1 = (const float*)d_in[9];
  const float* w2 = (const float*)d_in[10];
  const float* a0 = (const float*)d_in[11];
  const float* a1 = (const float*)d_in[12];
  const float* a2 = (const float*)d_in[13];
  const float* v0 = (const float*)d_in[14];
  const float* v1 = (const float*)d_in[15];
  const float* v2 = (const float*)d_in[16];
  const float* g1 = (const float*)d_in[17];
  const float* g2 = (const float*)d_in[18];
  const float* k_k = (const float*)d_in[19];
  const float* k_a = (const float*)d_in[20];
  const float* r_k = (const float*)d_in[21];
  const float* Wr = (const float*)d_in[22];
  const float* Wk = (const float*)d_in[23];
  const float* Wv = (const float*)d_in[24];
  const float* Wo = (const float*)d_in[25];
  const float* gamma = (const float*)d_in[26];
  const float* beta  = (const float*)d_in[27];

  char* ws = (char*)d_ws;
  size_t off = 0;
  auto alloc = [&](size_t bytes) -> void* {
    off = (off + 255) & ~(size_t)255;
    void* p = ws + off;
    off += bytes;
    return p;
  };

  const size_t MC = (size_t)Mrows * Cdim;
  const size_t CC = (size_t)Cdim * Cdim;

  // fragment-packed bf16 weights
  bf16_t* WrB = (bf16_t*)alloc(CC * 2);
  bf16_t* WkB = (bf16_t*)alloc(CC * 2);
  bf16_t* WvB = (bf16_t*)alloc(CC * 2);
  bf16_t* WoB = (bf16_t*)alloc(CC * 2);
  bf16_t* w1B = (bf16_t*)alloc((size_t)Cdim * 64 * 2);
  bf16_t* w2B = (bf16_t*)alloc((size_t)64 * Cdim * 2);
  bf16_t* a1B = (bf16_t*)alloc((size_t)Cdim * 64 * 2);
  bf16_t* a2B = (bf16_t*)alloc((size_t)64 * Cdim * 2);
  bf16_t* v1B = (bf16_t*)alloc((size_t)Cdim * 32 * 2);
  bf16_t* v2B = (bf16_t*)alloc((size_t)32 * Cdim * 2);
  bf16_t* g1B = (bf16_t*)alloc((size_t)Cdim * 128 * 2);
  bf16_t* g2B = (bf16_t*)alloc((size_t)128 * Cdim * 2);
  // bf16 mixed activations
  bf16_t* xrB = (bf16_t*)alloc(MC * 2);
  bf16_t* xwB = (bf16_t*)alloc(MC * 2);
  bf16_t* xkB = (bf16_t*)alloc(MC * 2);
  bf16_t* xvB = (bf16_t*)alloc(MC * 2);
  bf16_t* xaB = (bf16_t*)alloc(MC * 2);
  bf16_t* xgB = (bf16_t*)alloc(MC * 2);
  // f32 GEMM outputs
  float* rb   = (float*)alloc(MC * 4);
  float* kb   = (float*)alloc(MC * 4);
  float* vb   = (float*)alloc(MC * 4);
  float* wpre = (float*)alloc((size_t)Mrows * 64 * 4);
  float* apre = (float*)alloc((size_t)Mrows * 64 * 4);
  float* vpre = (float*)alloc((size_t)Mrows * 32 * 4);
  float* gpre = (float*)alloc((size_t)Mrows * 128 * 4);
  bf16_t* w2in = (bf16_t*)alloc((size_t)Mrows * 64 * 2);
  bf16_t* a2in = (bf16_t*)alloc((size_t)Mrows * 64 * 2);
  bf16_t* v2in = (bf16_t*)alloc((size_t)Mrows * 32 * 2);
  bf16_t* g2in = (bf16_t*)alloc((size_t)Mrows * 128 * 2);
  float* w2o  = (float*)alloc(MC * 4);   // -> decay (in place)
  float* a2o  = (float*)alloc(MC * 4);   // -> a sigmoid (in place)
  float* v2o  = (float*)alloc(MC * 4);
  float* gbuf = (float*)alloc(MC * 4);
  float* vnew = (float*)alloc(MC * 4);
  float* knew = (float*)alloc(MC * 4);
  float* kkb  = (float*)alloc(MC * 4);
  float* yb   = (float*)alloc(MC * 4);
  bf16_t* actB = (bf16_t*)alloc(MC * 2);

  const int TPB = 256;
  auto blks = [](size_t n, int t) { return (unsigned)((n + t - 1) / t); };

  // 1) weights -> fragment-packed bf16
  auto packb = [&](const float* src, bf16_t* dst, int N, int K) {
    pack_b_kernel<<<blks((size_t)N * K, TPB), TPB, 0, stream>>>(src, dst, N, K);
  };
  packb(Wr, WrB, Cdim, Cdim);
  packb(Wk, WkB, Cdim, Cdim);
  packb(Wv, WvB, Cdim, Cdim);
  packb(Wo, WoB, Cdim, Cdim);
  packb(w1, w1B, 64, Cdim);
  packb(w2, w2B, Cdim, 64);
  packb(a1, a1B, 64, Cdim);
  packb(a2, a2B, Cdim, 64);
  packb(v1, v1B, 32, Cdim);
  packb(v2, v2B, Cdim, 32);
  packb(g1, g1B, 128, Cdim);
  packb(g2, g2B, Cdim, 128);

  // 2) time-shift mixes
  mix6_kernel<<<blks(MC, TPB), TPB, 0, stream>>>(x, x_r, x_w, x_k, x_v, x_a, x_g,
                                                 xrB, xwB, xkB, xvB, xaB, xgB);

  // 3) GEMM dispatcher (WMMA)
  auto gemm = [&](const bf16_t* A, const bf16_t* Bp, float* Cm, int N, int K) {
    if (N % 64 == 0) {
      dim3 g(Mrows / 128, N / 64);
      wmma_gemm_bf16<4><<<g, 256, 0, stream>>>(A, Bp, Cm, Mrows, N, K);
    } else {
      dim3 g(Mrows / 128, N / 32);
      wmma_gemm_bf16<2><<<g, 256, 0, stream>>>(A, Bp, Cm, Mrows, N, K);
    }
  };
  gemm(xrB, WrB, rb, Cdim, Cdim);
  gemm(xkB, WkB, kb, Cdim, Cdim);
  gemm(xvB, WvB, vb, Cdim, Cdim);
  gemm(xwB, w1B, wpre, 64, Cdim);
  gemm(xaB, a1B, apre, 64, Cdim);
  gemm(xvB, v1B, vpre, 32, Cdim);
  gemm(xgB, g1B, gpre, 128, Cdim);

  // 4) activations between LoRA GEMMs
  act_kernel<<<blks((size_t)Mrows * 64, TPB), TPB, 0, stream>>>(wpre, w2in, Mrows * 64, 0);
  act_kernel<<<blks((size_t)Mrows * 64, TPB), TPB, 0, stream>>>(apre, a2in, Mrows * 64, 1);
  act_kernel<<<blks((size_t)Mrows * 32, TPB), TPB, 0, stream>>>(vpre, v2in, Mrows * 32, 1);
  act_kernel<<<blks((size_t)Mrows * 128, TPB), TPB, 0, stream>>>(gpre, g2in, Mrows * 128, 2);

  // 5) stage-2 GEMMs
  gemm(w2in, w2B, w2o, Cdim, 64);
  gemm(a2in, a2B, a2o, Cdim, 64);
  gemm(v2in, v2B, v2o, Cdim, 32);
  gemm(g2in, g2B, gbuf, Cdim, 128);

  // 6) decay / a / v-lerp / k*k_a
  post_kernel<<<blks(MC, TPB), TPB, 0, stream>>>(w2o, a2o, v2o, vb, v_first, kb,
                                                 w0, a0, v0, k_a,
                                                 w2o, a2o, vnew, knew);

  // 7) per-head l2norm
  kk_kernel<<<(unsigned)(Mrows * Hdim), 64, 0, stream>>>(kb, k_k, kkb);

  // 8) sequential state scan (async double-buffered)
  scan_kernel<<<Bdim * Hdim, 256, 0, stream>>>(rb, w2o, knew, vnew, kkb, a2o, yb);

  // 9) GroupNorm + residual + gate -> bf16
  gn_res_kernel<<<(unsigned)(Mrows * Hdim), 64, 0, stream>>>(yb, rb, knew, vnew, gbuf,
                                                             r_k, gamma, beta, actB);

  // 10) final projection -> d_out (f32)
  gemm(actB, WoB, (float*)d_out, Cdim, Cdim);

  (void)in_sizes; (void)n_in; (void)out_size; (void)ws_size;
}
